// TrackingHead_51616916963350
// MI455X (gfx1250) — matrix operand order repaired and checked by reading, not compiled, not run on hardware
//
#include <hip/hip_runtime.h>
#include <hip/hip_bf16.h>

#define CH    256
#define HGT   160
#define WID   160
#define HW    25600      // 160*160
#define BATCH 4
#define NBOX  512
#define KTOT  2304       // 256*9
#define REIDD 128
#define SW    161        // integral image side

typedef _Float16 v16h __attribute__((ext_vector_type(16)));
typedef float    v8f  __attribute__((ext_vector_type(8)));

union FragAB { v16h v; uint4 q[2]; };

// ---------------------------------------------------------------- cast f32->f16
__global__ void cast_f32_to_f16(const float4* __restrict__ src,
                                _Float16* __restrict__ dst, int n4) {
  int i = blockIdx.x * blockDim.x + threadIdx.x;
  if (i >= n4) return;
  float4 f = src[i];
  union { _Float16 h[4]; uint2 u; } r;
  r.h[0] = (_Float16)f.x; r.h[1] = (_Float16)f.y;
  r.h[2] = (_Float16)f.z; r.h[3] = (_Float16)f.w;
  ((uint2*)dst)[i] = r.u;
}

// -------------------------------------------------- fold BN into conv weights
__global__ void fold_weights(const float* __restrict__ cw, const float* __restrict__ cb,
                             const float* __restrict__ g,  const float* __restrict__ bt,
                             const float* __restrict__ mn, const float* __restrict__ vr,
                             _Float16* __restrict__ wf, float* __restrict__ bf, int total) {
  int i = blockIdx.x * blockDim.x + threadIdx.x;
  if (i >= total) return;
  int co = i / KTOT;                 // (layer*256 + c_out), 0..511
  float scale = g[co] * rsqrtf(vr[co] + 1e-5f);
  wf[i] = (_Float16)(cw[i] * scale);
  if ((i % KTOT) == 0)
    bf[co] = (cb[co] - mn[co]) * scale + bt[co];
}

// --------------------------------------- conv3x3 + BN + ReLU via implicit GEMM
// D[c_out][pixel] = sum_{k=ci*9+ky*3+kx} Wf[c_out][k] * X[ci][y+ky-1][x+kx-1]
__launch_bounds__(256, 1)
__global__ void conv_bn_relu_wmma(const _Float16* __restrict__ xin,
                                  const _Float16* __restrict__ wf,
                                  const float* __restrict__ bfv,
                                  _Float16* __restrict__ out16,
                                  float* __restrict__ out32,
                                  int doOut16, int doOut32) {
  __shared__ _Float16 Alds[128 * 40];   // weights tile [m][k], stride 40 (pad)
  __shared__ _Float16 Btlds[128 * 40];  // im2col tile transposed [n][k]
  __shared__ float    biasLds[128];

  const int tid  = threadIdx.x;
  const int b    = blockIdx.z;
  const int m0   = blockIdx.y * 128;
  const int p0   = blockIdx.x * 128;

  if (tid < 128) biasLds[tid] = bfv[m0 + tid];

  const int wv    = tid >> 5;        // wave 0..7
  const int lane  = tid & 31;
  const int lhalf = lane >> 4;       // 0/1
  const int lrow  = lane & 15;
  const int wm    = wv & 1;          // 2 waves along M (64 each)
  const int wn    = wv >> 1;         // 4 waves along N (32 each)

  // loader roles
  const int kLoc = tid >> 3;         // 0..31  (B tile k)
  const int nGrp = (tid & 7) << 4;   // 0..112 (B tile n group of 16)
  const int am   = tid >> 1;         // 0..127 (A tile row)
  const int apart= tid & 1;          // which 16 of 32 k's

  // Pixel group: 16-aligned and 160 % 16 == 0 => never crosses an image row.
  const int pGrp = p0 + nGrp;
  const int yG   = pGrp / WID;
  const int xG   = pGrp - yG * WID;          // multiple of 16
  _Float16* const dstcol = &Btlds[nGrp * 40 + kLoc];

  // incremental decode of k = k0 + kLoc into (ci, r9); advance by 32 per step
  int ci = kLoc / 9;
  int r9 = kLoc - ci * 9;
  int chanOff = ci * HW;                      // element offset of channel plane

  // A-tile async-DMA staging: per-lane global source and LDS destination
  const _Float16* aSrc = wf + (size_t)(m0 + am) * KTOT + apart * 16;
  const unsigned  aLds = (unsigned)(size_t)&Alds[am * 40 + apart * 16];

  v8f acc[4][2];
  #pragma unroll
  for (int mi = 0; mi < 4; ++mi)
    #pragma unroll
    for (int ni = 0; ni < 2; ++ni)
      #pragma unroll
      for (int r = 0; r < 8; ++r) acc[mi][ni][r] = 0.f;

  const _Float16* xb = xin + (size_t)b * CH * HW;

  for (int k0 = 0; k0 < KTOT; k0 += 32) {
    __syncthreads();
    // ---- stage A (weights) tile: 128 x 32 f16 via async global->LDS DMA
    // (ASYNCcnt-tracked; drained before the consumer barrier below)
    {
      const _Float16* srcp = aSrc + k0;
      asm volatile(
          "global_load_async_to_lds_b128 %0, %1, off\n\t"
          "global_load_async_to_lds_b128 %0, %1, off offset:16"
          :
          : "v"(aLds), "v"(srcp)
          : "memory");
    }
    // ---- stage B (im2col) tile transposed: [n 0..127][k 0..31]
    {
      const int dyr = r9 / 3;
      const int dy  = dyr - 1;
      const int dx  = r9 - dyr * 3 - 1;
      const int sy  = yG + dy;
      const int sxb = xG + dx;
      if ((unsigned)sy < (unsigned)HGT) {
        const _Float16* rp = xb + chanOff + sy * WID;
        // only the two edge elements can be out of bounds in x (dx = +/-1)
        _Float16 v0  = ((unsigned)sxb < (unsigned)WID) ? rp[sxb] : (_Float16)0.f;
        dstcol[0] = v0;
        #pragma unroll
        for (int j = 1; j < 15; ++j)
          dstcol[j * 40] = rp[sxb + j];
        _Float16 v15 = ((unsigned)(sxb + 15) < (unsigned)WID) ? rp[sxb + 15]
                                                              : (_Float16)0.f;
        dstcol[15 * 40] = v15;
      } else {
        #pragma unroll
        for (int j = 0; j < 16; ++j)
          dstcol[j * 40] = (_Float16)0.f;
      }
      // advance k by 32: 32 = 3*9 + 5
      chanOff += 3 * HW;
      r9 += 5;
      if (r9 >= 9) { r9 -= 9; chanOff += HW; }
    }
    // ensure the async DMA landed in LDS before any wave consumes the tile
    asm volatile("s_wait_asynccnt 0" ::: "memory");
    __syncthreads();

    // ---- fragment loads (contiguous b128 pairs thanks to LDS arrangement)
    FragAB af[4], bfr[2];
    #pragma unroll
    for (int mi = 0; mi < 4; ++mi) {
      const _Float16* base = &Alds[(wm * 64 + mi * 16 + lrow) * 40];
      af[mi].q[0] = *(const uint4*)(base + lhalf * 16);       // K = half*16 + 0..7
      af[mi].q[1] = *(const uint4*)(base + lhalf * 16 + 8);   // K = half*16 + 8..15
    }
    #pragma unroll
    for (int ni = 0; ni < 2; ++ni) {
      const _Float16* base = &Btlds[(wn * 32 + ni * 16 + lrow) * 40 + lhalf * 16];
      bfr[ni].q[0] = ((const uint4*)base)[0];                 // K = half*16 + 0..15
      bfr[ni].q[1] = ((const uint4*)base)[1];
    }
    #pragma unroll
    for (int mi = 0; mi < 4; ++mi)
      #pragma unroll
      for (int ni = 0; ni < 2; ++ni)
        acc[mi][ni] = __builtin_amdgcn_wmma_f32_16x16x32_f16(
            false, af[mi].v, false, bfr[ni].v, (short)0, acc[mi][ni], false, false);
  }

  // ---- epilogue: bias + ReLU, C/D layout: VGPR r -> M = half*8+r, N = lane%16
  #pragma unroll
  for (int mi = 0; mi < 4; ++mi) {
    #pragma unroll
    for (int ni = 0; ni < 2; ++ni) {
      const int p = p0 + wn * 32 + ni * 16 + lrow;
      #pragma unroll
      for (int r = 0; r < 8; ++r) {
        const int coL = wm * 64 + mi * 16 + lhalf * 8 + r;
        float v = acc[mi][ni][r] + biasLds[coL];
        v = fmaxf(v, 0.f);
        const size_t off = (size_t)(b * CH + m0 + coL) * HW + p;
        if (doOut16) out16[off] = (_Float16)v;
        if (doOut32) out32[off] = v;
      }
    }
  }
}

// ------------------------------------------- integral image: row scan in place
__global__ void row_scan_inplace(float* __restrict__ data, int nrows) {
  int gwave = (blockIdx.x * blockDim.x + threadIdx.x) >> 5;
  int lane  = threadIdx.x & 31;
  if (gwave >= nrows) return;
  float* row = data + (size_t)gwave * WID;
  float carry = 0.f;
  #pragma unroll
  for (int c = 0; c < 5; ++c) {
    float v = row[c * 32 + lane];
    #pragma unroll
    for (int off = 1; off < 32; off <<= 1) {
      float u = __shfl_up(v, off, 32);
      if (lane >= off) v += u;
    }
    v += carry;
    row[c * 32 + lane] = v;
    carry = __shfl(v, 31, 32);
  }
}

// --------------- transpose NCHW -> padded NHWC: T[b][y+1][x+1][c] = rowsum
__launch_bounds__(256)
__global__ void transpose_pad(const float* __restrict__ src, float* __restrict__ T) {
  __shared__ float tile[32][33];
  const int tx = threadIdx.x, ty = threadIdx.y;
  const int xt = blockIdx.x % 5, ct = blockIdx.x / 5;
  const int y  = blockIdx.y,     b  = blockIdx.z;
  const int x0 = xt * 32, c0 = ct * 32;
  #pragma unroll
  for (int k = 0; k < 4; ++k) {
    int c = c0 + ty + k * 8;
    tile[ty + k * 8][tx] = src[((size_t)(b * CH + c) * HGT + y) * WID + x0 + tx];
  }
  __syncthreads();
  #pragma unroll
  for (int k = 0; k < 4; ++k) {
    int xx = x0 + ty + k * 8;
    T[(((size_t)b * SW + (y + 1)) * SW + (xx + 1)) * CH + c0 + tx] = tile[tx][ty + k * 8];
  }
}

__global__ void zero_borders(float* __restrict__ T) {
  int i = blockIdx.x * blockDim.x + threadIdx.x;
  const int tot = BATCH * SW * CH;
  if (i >= tot) return;
  int c = i % CH; int q = (i / CH) % SW; int b = i / (CH * SW);
  T[(((size_t)b * SW + 0) * SW + q) * CH + c] = 0.f;   // yy = 0 row
  T[(((size_t)b * SW + q) * SW + 0) * CH + c] = 0.f;   // xx = 0 col
}

__global__ void col_scan_inplace(float* __restrict__ T) {
  int i = blockIdx.x * blockDim.x + threadIdx.x;
  const int tot = BATCH * SW * CH;
  if (i >= tot) return;
  int c = i % CH; int xx = (i / CH) % SW; int b = i / (CH * SW);
  size_t base = (((size_t)b * SW) * SW + xx) * CH + c;
  const size_t stride = (size_t)SW * CH;
  float acc = 0.f;
  for (int yy = 1; yy <= HGT; ++yy) {
    size_t idx = base + (size_t)yy * stride;
    acc += T[idx];
    T[idx] = acc;
  }
}

// ----------------------------------------- ROI pooling via integral image
__launch_bounds__(256)
__global__ void roi_pool(const float* __restrict__ T, const float* __restrict__ boxes,
                         float* __restrict__ pooled, int* __restrict__ valid) {
  const int bn = blockIdx.x;
  const int c  = threadIdx.x;
  const int b  = bn / NBOX;
  const float4 bx = ((const float4*)boxes)[bn];
  int x1 = min(max((int)floorf(bx.x * (float)WID), 0), WID);
  int y1 = min(max((int)floorf(bx.y * (float)HGT), 0), HGT);
  int x2 = min(max((int)floorf(bx.z * (float)WID), 0), WID);
  int y2 = min(max((int)floorf(bx.w * (float)HGT), 0), HGT);
  if (c == 0) valid[bn] = (x2 > x1 && y2 > y1) ? 1 : 0;
  int hl = max(y2 - y1, 1), wl = max(x2 - x1, 1);
  int hs[7], he[7], ws[7], we[7];
  #pragma unroll
  for (int i = 0; i < 7; ++i) {
    hs[i] = y1 + (i * hl) / 7;  he[i] = y1 + ((i + 1) * hl + 6) / 7;
    ws[i] = x1 + (i * wl) / 7;  we[i] = x1 + ((i + 1) * wl + 6) / 7;
  }
  const float* Sb = T + (size_t)b * SW * SW * CH;
  float sum = 0.f;
  #pragma unroll
  for (int i = 0; i < 7; ++i) {
    #pragma unroll
    for (int j = 0; j < 7; ++j) {
      float s = Sb[((size_t)he[i] * SW + we[j]) * CH + c]
              - Sb[((size_t)hs[i] * SW + we[j]) * CH + c]
              - Sb[((size_t)he[i] * SW + ws[j]) * CH + c]
              + Sb[((size_t)hs[i] * SW + ws[j]) * CH + c];
      float area = (float)((he[i] - hs[i]) * (we[j] - ws[j]));
      sum += s / area;
    }
  }
  pooled[(size_t)bn * CH + c] = sum * (1.f / 49.f);
}

// ------------------------------------------- MLP + mask + L2 normalize
__launch_bounds__(128)
__global__ void mlp_norm(const float* __restrict__ pooled, const int* __restrict__ valid,
                         const float* __restrict__ w1, const float* __restrict__ b1,
                         const float* __restrict__ w2, const float* __restrict__ b2,
                         float* __restrict__ out) {
  const int bn = blockIdx.x, tid = threadIdx.x;
  __shared__ float pool[CH];
  __shared__ float hbuf[REIDD];
  __shared__ float red[4];
  pool[tid]        = pooled[(size_t)bn * CH + tid];
  pool[tid + 128]  = pooled[(size_t)bn * CH + tid + 128];
  __syncthreads();
  float s = b1[tid];
  for (int k = 0; k < CH; ++k) s += w1[tid * CH + k] * pool[k];
  hbuf[tid] = fmaxf(s, 0.f);
  __syncthreads();
  float f = b2[tid];
  for (int k = 0; k < REIDD; ++k) f += w2[tid * REIDD + k] * hbuf[k];
  if (!valid[bn]) f = 0.f;
  float sq = f * f;
  for (int off = 16; off > 0; off >>= 1) sq += __shfl_down(sq, off, 32);
  if ((tid & 31) == 0) red[tid >> 5] = sq;
  __syncthreads();
  float tot = red[0] + red[1] + red[2] + red[3];
  float nrm = fmaxf(sqrtf(tot), 1e-12f);
  out[(size_t)bn * REIDD + tid] = f / nrm;
}

// =============================================================== launcher
extern "C" void kernel_launch(void* const* d_in, const int* in_sizes, int n_in,
                              void* d_out, int out_size, void* d_ws, size_t ws_size,
                              hipStream_t stream) {
  const float* x      = (const float*)d_in[0];
  const float* bboxes = (const float*)d_in[1];
  const float* conv_w = (const float*)d_in[2];
  const float* conv_b = (const float*)d_in[3];
  const float* bn_g   = (const float*)d_in[4];
  const float* bn_b   = (const float*)d_in[5];
  const float* bn_m   = (const float*)d_in[6];
  const float* bn_v   = (const float*)d_in[7];
  const float* w1     = (const float*)d_in[8];
  const float* b1     = (const float*)d_in[9];
  const float* w2     = (const float*)d_in[10];
  const float* b2     = (const float*)d_in[11];

  char* ws = (char*)d_ws;
  size_t off = 0;
  auto alloc = [&](size_t bytes) -> char* {
    char* p = ws + off;
    off = (off + bytes + 255) & ~(size_t)255;
    return p;
  };
  _Float16* xhA    = (_Float16*)alloc((size_t)BATCH * CH * HW * 2);
  _Float16* xhB    = (_Float16*)alloc((size_t)BATCH * CH * HW * 2);
  _Float16* wf     = (_Float16*)alloc((size_t)2 * CH * KTOT * 2);
  float*    bf     = (float*)  alloc((size_t)2 * CH * 4);
  float*    conv32 = (float*)  alloc((size_t)BATCH * CH * HW * 4);
  float*    T      = (float*)  alloc((size_t)BATCH * SW * SW * CH * 4);
  float*    pooled = (float*)  alloc((size_t)BATCH * NBOX * CH * 4);
  int*      valid  = (int*)    alloc((size_t)BATCH * NBOX * 4);

  // 1) cast input to f16
  int n4 = BATCH * CH * HW / 4;
  cast_f32_to_f16<<<(n4 + 255) / 256, 256, 0, stream>>>((const float4*)x, xhA, n4);

  // 2) fold BN into weights/bias
  int wtot = 2 * CH * KTOT;
  fold_weights<<<(wtot + 255) / 256, 256, 0, stream>>>(conv_w, conv_b, bn_g, bn_b,
                                                       bn_m, bn_v, wf, bf, wtot);

  // 3) conv layer 1 (f16 out), 4) conv layer 2 (f32 out)
  dim3 cg(HW / 128, CH / 128, BATCH);   // 200 x 2 x 4
  conv_bn_relu_wmma<<<cg, 256, 0, stream>>>(xhA, wf, bf, xhB, conv32, 1, 0);
  conv_bn_relu_wmma<<<cg, 256, 0, stream>>>(xhB, wf + (size_t)CH * KTOT, bf + CH,
                                            xhA, conv32, 0, 1);

  // 5) integral image: row scan (in place), transpose to padded NHWC, borders, col scan
  int nrows = BATCH * CH * HGT;
  row_scan_inplace<<<(nrows * 32 + 255) / 256, 256, 0, stream>>>(conv32, nrows);
  dim3 tg(40, HGT, BATCH), tb(32, 8);
  transpose_pad<<<tg, tb, 0, stream>>>(conv32, T);
  int btot = BATCH * SW * CH;
  zero_borders<<<(btot + 255) / 256, 256, 0, stream>>>(T);
  col_scan_inplace<<<(btot + 255) / 256, 256, 0, stream>>>(T);

  // 6) ROI pool, 7) MLP + normalize
  roi_pool<<<BATCH * NBOX, 256, 0, stream>>>(T, bboxes, pooled, valid);
  mlp_norm<<<BATCH * NBOX, 128, 0, stream>>>(pooled, valid, w1, b1, w2, b2,
                                             (float*)d_out);
}
